// MultiHeadAttention_70248485093572
// MI455X (gfx1250) — matrix operand order repaired and checked by reading, not compiled, not run on hardware
//
#include <hip/hip_runtime.h>
#include <hip/hip_bf16.h>

// Problem constants (fixed by the reference)
#define Bq 2
#define Sq 4096
#define Dq 512
#define Hq 8
#define DKq 64
#define BSq (Bq * Sq)   // 8192

typedef __attribute__((ext_vector_type(16))) __bf16 v16bf;
typedef __attribute__((ext_vector_type(8)))  float  v8f;

union FragBF {
    v16bf v;
    uint4 u4[2];
};

__device__ __forceinline__ unsigned short f2bf(float f) {
    union { float f; unsigned int u; } c;
    c.f = f;
    unsigned int u = c.u;
    unsigned int r = (u + 0x7FFFu + ((u >> 16) & 1u)) >> 16;  // RNE
    return (unsigned short)r;
}

// ---------------------------------------------------------------------------
// fp32 -> bf16 conversion
// ---------------------------------------------------------------------------
__global__ void cvt_bf16_kernel(const float* __restrict__ in,
                                unsigned short* __restrict__ out, int n) {
    int i = blockIdx.x * blockDim.x + threadIdx.x;
    if (i < n) out[i] = f2bf(in[i]);
}

// ---------------------------------------------------------------------------
// Fragment loaders (row-major bf16 memory, contiguous-k VGPR layouts, wave32)
// ---------------------------------------------------------------------------
// A-matrix 16x32: lane l -> row (l&15); K elements h*16 .. +15 contiguous
__device__ __forceinline__ FragBF load_frag_A(const unsigned short* base, int ld,
                                              int row, int col, int lane) {
    FragBF f;
    const unsigned short* p = base + (size_t)(row + (lane & 15)) * ld + col;
    int h = lane >> 4;
    f.u4[0] = *(const uint4*)(p + h * 16);
    f.u4[1] = *(const uint4*)(p + h * 16 + 8);
    return f;
}

// B-matrix 32x16 taken from row-major [N][K] (i.e. computes A x W^T):
// lane l -> column (l&15); K elements (l>>4)*16 .. +15 contiguous.
__device__ __forceinline__ FragBF load_frag_B(const unsigned short* base, int ld,
                                              int nrow, int col, int lane) {
    FragBF f;
    const unsigned short* p =
        base + (size_t)(nrow + (lane & 15)) * ld + col + (lane >> 4) * 16;
    f.u4[0] = *(const uint4*)(p);
    f.u4[1] = *(const uint4*)(p + 8);
    return f;
}

__device__ __forceinline__ v8f wmma_bf16(const FragBF& a, const FragBF& b, v8f c) {
    return __builtin_amdgcn_wmma_f32_16x16x32_bf16(
        false, a.v, false, b.v, (short)0, c, false, false);
}

// ---------------------------------------------------------------------------
// GEMM: C[M x 512] = A[M x 512] (bf16) * W[512 x 512]^T (bf16, torch layout)
// mode 0: store f32 row-major -> Cf
// mode 1: store bf16 into per-head [B*H][S][64]        (Q, K)
// mode 2: store bf16 into transposed [B*H][64][S]      (V)
// Block = 256 threads (8 waves); wave tile 64x64; block tile 256x128.
// ---------------------------------------------------------------------------
__global__ __launch_bounds__(256) void gemm_bf16_kernel(
    const unsigned short* __restrict__ A, const unsigned short* __restrict__ W,
    float* __restrict__ Cf, unsigned short* __restrict__ Cbf, int mode) {
    const int K = Dq, N = Dq;
    const int lane = threadIdx.x & 31;
    const int wave = threadIdx.x >> 5;
    const int rowBase = blockIdx.x * 256 + (wave >> 1) * 64;
    const int colBase = blockIdx.y * 128 + (wave & 1) * 64;

    v8f acc[4][4];
#pragma unroll
    for (int i = 0; i < 4; ++i)
#pragma unroll
        for (int j = 0; j < 4; ++j) acc[i][j] = (v8f)(0.0f);

    for (int kk = 0; kk < K; kk += 32) {
        FragBF a[4], b[4];
#pragma unroll
        for (int i = 0; i < 4; ++i)
            a[i] = load_frag_A(A, K, rowBase + i * 16, kk, lane);
#pragma unroll
        for (int j = 0; j < 4; ++j)
            b[j] = load_frag_B(W, K, colBase + j * 16, kk, lane);
#pragma unroll
        for (int i = 0; i < 4; ++i)
#pragma unroll
            for (int j = 0; j < 4; ++j)
                acc[i][j] = wmma_bf16(a[i], b[j], acc[i][j]);
    }

    const int hf = lane >> 4, nn = lane & 15;
#pragma unroll
    for (int i = 0; i < 4; ++i)
#pragma unroll
        for (int j = 0; j < 4; ++j)
#pragma unroll
            for (int r = 0; r < 8; ++r) {
                float v = acc[i][j][r];
                int row = rowBase + i * 16 + r + 8 * hf;  // 0..8191
                int col = colBase + j * 16 + nn;          // 0..511
                if (mode == 0) {
                    Cf[(size_t)row * N + col] = v;
                } else {
                    unsigned short bv = f2bf(v);
                    int bb = row >> 12, ss = row & (Sq - 1);
                    int hd = col >> 6, dk = col & (DKq - 1);
                    if (mode == 1)
                        Cbf[(((size_t)(bb * Hq + hd)) * Sq + ss) * DKq + dk] = bv;
                    else
                        Cbf[(((size_t)(bb * Hq + hd)) * DKq + dk) * Sq + ss] = bv;
                }
            }
}

// ---------------------------------------------------------------------------
// Flash attention: per wave, one 16-query tile of one (b, head).
// Qh, Kh: [B*H][S][64] bf16 ; Vt: [B*H][64][S] bf16 ; Ob: [B*S][512] bf16.
// Keys processed in chunks of 32 with online softmax; scale = 1/sqrt(64).
// ---------------------------------------------------------------------------
__global__ __launch_bounds__(128) void attn_kernel(
    const unsigned short* __restrict__ Qh, const unsigned short* __restrict__ Kh,
    const unsigned short* __restrict__ Vt, unsigned short* __restrict__ Ob) {
    __shared__ unsigned short plds[4][16][32];  // P staging per wave (1 KB each)

    const int lane = threadIdx.x & 31;
    const int wave = threadIdx.x >> 5;
    const int bh = blockIdx.x;                  // 0..15
    const int qtile = blockIdx.y * 4 + wave;    // 0..255
    const int qrow = qtile * 16;
    const int m = lane & 15, h = lane >> 4;

    const unsigned short* Qb = Qh + (size_t)bh * Sq * DKq;
    const unsigned short* Kb = Kh + (size_t)bh * Sq * DKq;
    const unsigned short* Vb = Vt + (size_t)bh * DKq * Sq;

    FragBF qa0 = load_frag_A(Qb, DKq, qrow, 0, lane);
    FragBF qa1 = load_frag_A(Qb, DKq, qrow, 32, lane);

    v8f acc[4];
#pragma unroll
    for (int f = 0; f < 4; ++f) acc[f] = (v8f)(0.0f);
    float mrow[8], lrow[8];
#pragma unroll
    for (int r = 0; r < 8; ++r) { mrow[r] = -1.0e30f; lrow[r] = 0.0f; }

    const float L2E = 1.4426950408889634f;

    for (int kc = 0; kc < Sq; kc += 32) {
        // scores: two 16x16 f32 fragments over this 32-key chunk
        v8f sc0 = (v8f)(0.0f), sc1 = (v8f)(0.0f);
        FragBF kb;
        kb = load_frag_B(Kb, DKq, kc, 0, lane);       sc0 = wmma_bf16(qa0, kb, sc0);
        kb = load_frag_B(Kb, DKq, kc, 32, lane);      sc0 = wmma_bf16(qa1, kb, sc0);
        kb = load_frag_B(Kb, DKq, kc + 16, 0, lane);  sc1 = wmma_bf16(qa0, kb, sc1);
        kb = load_frag_B(Kb, DKq, kc + 16, 32, lane); sc1 = wmma_bf16(qa1, kb, sc1);

        // online softmax per row (rows live in VGPR index; columns across lanes)
        float e0[8], e1[8];
#pragma unroll
        for (int r = 0; r < 8; ++r) {
            float s0 = sc0[r] * 0.125f;
            float s1 = sc1[r] * 0.125f;
            float cm = fmaxf(s0, s1);
            cm = fmaxf(cm, __shfl_xor(cm, 1, 32));
            cm = fmaxf(cm, __shfl_xor(cm, 2, 32));
            cm = fmaxf(cm, __shfl_xor(cm, 4, 32));
            cm = fmaxf(cm, __shfl_xor(cm, 8, 32));
            float mn = fmaxf(mrow[r], cm);
            float alpha = __builtin_amdgcn_exp2f((mrow[r] - mn) * L2E);
            e0[r] = __builtin_amdgcn_exp2f((s0 - mn) * L2E);
            e1[r] = __builtin_amdgcn_exp2f((s1 - mn) * L2E);
            float rs = e0[r] + e1[r];
            rs += __shfl_xor(rs, 1, 32);
            rs += __shfl_xor(rs, 2, 32);
            rs += __shfl_xor(rs, 4, 32);
            rs += __shfl_xor(rs, 8, 32);
            lrow[r] = lrow[r] * alpha + rs;
            mrow[r] = mn;
#pragma unroll
            for (int f = 0; f < 4; ++f) acc[f][r] *= alpha;
        }

        // stage P (C layout -> LDS), transpose-read as A fragment
#pragma unroll
        for (int r = 0; r < 8; ++r) {
            plds[wave][r + 8 * h][m] = f2bf(e0[r]);
            plds[wave][r + 8 * h][m + 16] = f2bf(e1[r]);
        }
        asm volatile("s_wait_dscnt 0x0" ::: "memory");
        FragBF pa;
        pa.u4[0] = *(const uint4*)&plds[wave][m][h * 16];
        pa.u4[1] = *(const uint4*)&plds[wave][m][h * 16 + 8];

        // acc += P(16x32) * V(32x64): V^T layout gives contiguous B loads
#pragma unroll
        for (int f = 0; f < 4; ++f) {
            FragBF vb;
            const unsigned short* vp =
                Vb + (size_t)(f * 16 + m) * Sq + kc + h * 16;
            vb.u4[0] = *(const uint4*)(vp);
            vb.u4[1] = *(const uint4*)(vp + 8);
            acc[f] = wmma_bf16(pa, vb, acc[f]);
        }
    }

    // epilogue: normalize and scatter into [B*S][512] bf16 (merge heads)
    const int bb = bh >> 3, head = bh & 7;
#pragma unroll
    for (int r = 0; r < 8; ++r) {
        float inv = __builtin_amdgcn_rcpf(lrow[r]);
        int s = qrow + r + 8 * h;
#pragma unroll
        for (int f = 0; f < 4; ++f) {
            int col = head * DKq + f * 16 + m;
            Ob[((size_t)(bb * Sq + s)) * Dq + col] = f2bf(acc[f][r] * inv);
        }
    }
}

// ---------------------------------------------------------------------------
// Host-side orchestration
// ---------------------------------------------------------------------------
extern "C" void kernel_launch(void* const* d_in, const int* in_sizes, int n_in,
                              void* d_out, int out_size, void* d_ws, size_t ws_size,
                              hipStream_t stream) {
    const float* q_in = (const float*)d_in[0];
    const float* k_in = (const float*)d_in[1];
    const float* v_in = (const float*)d_in[2];
    const float* Wq = (const float*)d_in[3];
    const float* Wk = (const float*)d_in[4];
    const float* Wv = (const float*)d_in[5];
    const float* Wo = (const float*)d_in[6];
    float* out = (float*)d_out;

    // Workspace layout (all bf16 / unsigned short elements)
    const size_t N_IN = (size_t)BSq * Dq;   // 4,194,304
    const size_t N_W = (size_t)Dq * Dq;     // 262,144
    unsigned short* ws = (unsigned short*)d_ws;
    unsigned short* qbf = ws;
    unsigned short* kbf = qbf + N_IN;
    unsigned short* vbf = kbf + N_IN;
    unsigned short* wqb = vbf + N_IN;
    unsigned short* wkb = wqb + N_W;
    unsigned short* wvb = wkb + N_W;
    unsigned short* wob = wvb + N_W;
    unsigned short* Qh = wob + N_W;         // [B*H][S][64]
    unsigned short* Kh = Qh + N_IN;
    unsigned short* Vt = Kh + N_IN;         // [B*H][64][S]
    unsigned short* attnb = Vt + N_IN;      // [B*S][512]

    // 1) fp32 -> bf16 conversions
    cvt_bf16_kernel<<<(int)(N_IN / 256), 256, 0, stream>>>(q_in, qbf, (int)N_IN);
    cvt_bf16_kernel<<<(int)(N_IN / 256), 256, 0, stream>>>(k_in, kbf, (int)N_IN);
    cvt_bf16_kernel<<<(int)(N_IN / 256), 256, 0, stream>>>(v_in, vbf, (int)N_IN);
    cvt_bf16_kernel<<<(int)(N_W / 256), 256, 0, stream>>>(Wq, wqb, (int)N_W);
    cvt_bf16_kernel<<<(int)(N_W / 256), 256, 0, stream>>>(Wk, wkb, (int)N_W);
    cvt_bf16_kernel<<<(int)(N_W / 256), 256, 0, stream>>>(Wv, wvb, (int)N_W);
    cvt_bf16_kernel<<<(int)(N_W / 256), 256, 0, stream>>>(Wo, wob, (int)N_W);

    // 2) projections (WMMA GEMMs): block tile 256x128 over M=8192, N=512
    dim3 gg(BSq / 256, Dq / 128);
    gemm_bf16_kernel<<<gg, 256, 0, stream>>>(qbf, wqb, nullptr, Qh, 1);
    gemm_bf16_kernel<<<gg, 256, 0, stream>>>(kbf, wkb, nullptr, Kh, 1);
    gemm_bf16_kernel<<<gg, 256, 0, stream>>>(vbf, wvb, nullptr, Vt, 2);

    // 3) flash attention (WMMA + online softmax)
    attn_kernel<<<dim3(Bq * Hq, Sq / (16 * 4)), 128, 0, stream>>>(Qh, Kh, Vt, attnb);

    // 4) output projection -> fp32 d_out
    gemm_bf16_kernel<<<gg, 256, 0, stream>>>(attnb, wob, out, nullptr, 0);
}
